// SimpleSplitGMM_25262997635213
// MI455X (gfx1250) — compile-verified, matrix operand order checked by reading, not run on hardware
//
#include <hip/hip_runtime.h>
#include <hip/hip_bf16.h>

// ---------------------------------------------------------------------------
// SimpleSplitGMM on MI455X (gfx1250), v6:
//  - full 128x384 A block (z, bf16) resident in LDS (frees afrag registers)
//  - all 8 B fragments batch-loaded per K-step, then 8 back-to-back WMMAs
//  - means pre-converted to bf16 once; B staging = async global->LDS (ASYNCcnt)
//  - packed v_cvt_pk_bf16_f32 conversions, 8-byte LDS stores
//  - double-buffered B tiles, 1 barrier per K-step
//  - non-temporal logit stores
// GEMM core: v_wmma_f32_16x16x32_bf16, f32 accumulate, fused epilogue
// (logit store + argmax + label-gather + loss reduction).
// ---------------------------------------------------------------------------

#define NS        16384
#define TOTAL_DIM 3072
#define NUM_ATTR  8
#define NC        512
#define KD        384
#define BM        128
#define BN        128
#define TK        32
#define KSTEPS    (KD / TK)          // 12
#define BSTRIDE   (TK + 8)           // 40 bf16 = 80B: 16B-aligned, bank-safe

static constexpr float LOG_2PI = 1.8378770664093453f;

typedef __bf16 bf16_t;
typedef __attribute__((ext_vector_type(16))) __bf16 v16bf;
typedef __attribute__((ext_vector_type(8)))  __bf16 v8bf;
typedef __attribute__((ext_vector_type(2)))  __bf16 v2bf;
typedef __attribute__((ext_vector_type(8)))  float  v8f;

// Types matching the async-LDS builtin prototype (confirmed by hipcc):
//   arg0: int4 addrspace(1)*, arg1: int4 addrspace(3)*
typedef int vsi4 __attribute__((vector_size(16)));
typedef __attribute__((address_space(1))) vsi4 as1_vsi4;
typedef __attribute__((address_space(3))) vsi4 as3_vsi4;

#if __has_builtin(__builtin_amdgcn_global_load_async_to_lds_b128) && \
    __has_builtin(__builtin_amdgcn_s_wait_asynccnt)
#define HAVE_ASYNC_LDS 1
#else
#define HAVE_ASYNC_LDS 0
#endif

__device__ __forceinline__ bf16_t f32_to_bf16(float f) {
  union { float f; unsigned u; } c; c.f = f;
  unsigned r = (c.u + 0x7FFFu + ((c.u >> 16) & 1u)) >> 16;   // RNE
  union { unsigned short s; bf16_t b; } o; o.s = (unsigned short)r;
  return o.b;
}

// Pair of f32 -> packed bf16x2 (one VALU op via HW packed cvt).
__device__ __forceinline__ v2bf cvt2(float x, float y) {
#if __has_builtin(__builtin_amdgcn_cvt_pk_bf16_f32)
  return __builtin_amdgcn_cvt_pk_bf16_f32(x, y);
#elif defined(__gfx1250__)
  v2bf o;
  asm("v_cvt_pk_bf16_f32 %0, %1, %2" : "=v"(o) : "v"(x), "v"(y));
  return o;
#else
  v2bf o; o[0] = f32_to_bf16(x); o[1] = f32_to_bf16(y); return o;
#endif
}

// Convert float4 -> 4 bf16 and store as one 8-byte chunk.
__device__ __forceinline__ void cvt4_store(bf16_t* dst, float4 v) {
  union { v2bf h[2]; unsigned long long u; } pk;
  pk.h[0] = cvt2(v.x, v.y);
  pk.h[1] = cvt2(v.z, v.w);
  *reinterpret_cast<unsigned long long*>(dst) = pk.u;
}

__device__ __forceinline__ void wait_async_staging() {
#if HAVE_ASYNC_LDS
  __builtin_amdgcn_s_wait_asynccnt(0);
#endif
}

__global__ void gmm_init_kernel(float* out_loss) {
  if (threadIdx.x == 0 && blockIdx.x == 0) out_loss[0] = 0.0f;
}

// means f32 -> bf16 (1.57M elements, one pass)
__global__ void gmm_cvt_means_kernel(const float* __restrict__ means,
                                     bf16_t* __restrict__ mbf) {
  size_t idx = (size_t)blockIdx.x * blockDim.x + threadIdx.x;  // float4 index
  float4 v = *reinterpret_cast<const float4*>(means + idx * 4);
  union { v2bf h[2]; unsigned long long u; } pk;
  pk.h[0] = cvt2(v.x, v.y);
  pk.h[1] = cvt2(v.z, v.w);
  *reinterpret_cast<unsigned long long*>(mbf + idx * 4) = pk.u;
}

// msq[a*NC + c] = sum_d means[a][c][d]^2 (f32 source for accuracy)
__global__ void gmm_msq_kernel(const float* __restrict__ means,
                               float* __restrict__ msq) {
  int i = blockIdx.x * blockDim.x + threadIdx.x;
  const float* m = means + (size_t)i * KD;
  float s = 0.0f;
  for (int d = 0; d < KD; d += 4) {
    float4 v = *reinterpret_cast<const float4*>(m + d);
    s += v.x * v.x + v.y * v.y + v.z * v.z + v.w * v.w;
  }
  msq[i] = s;
}

// Stage one 128x32 bf16 tile of pre-converted means into LDS (async copy).
__device__ __forceinline__ void stage_B(const bf16_t* __restrict__ mbf_attr,
                                        int tid, bf16_t* buf, int cb, int ks) {
  #pragma unroll
  for (int it = 0; it < 2; ++it) {
    int idx = it * 256 + tid;            // 512 x 16B chunks
    int c   = idx >> 2;                  // col 0..127
    int q   = idx & 3;                   // 16B quarter of the 64B row chunk
    const bf16_t* src = mbf_attr + (size_t)(cb * BN + c) * KD + ks * TK + q * 8;
    bf16_t* dst = buf + c * BSTRIDE + q * 8;
#if HAVE_ASYNC_LDS
    __builtin_amdgcn_global_load_async_to_lds_b128(
        (as1_vsi4*)src, (as3_vsi4*)dst, 0, 0);
#else
    *reinterpret_cast<uint4*>(dst) = *reinterpret_cast<const uint4*>(src);
#endif
  }
}

__device__ __forceinline__ v16bf frag_from_lds(const bf16_t* base) {
  v8bf lo = *reinterpret_cast<const v8bf*>(base);
  v8bf hi = *reinterpret_cast<const v8bf*>(base + 16);
  return __builtin_shufflevector(lo, hi,
      0, 1, 2, 3, 4, 5, 6, 7, 8, 9, 10, 11, 12, 13, 14, 15);
}

__global__ void __launch_bounds__(256)
gmm_main_kernel(const float* __restrict__ z,
                const float* __restrict__ sldj,
                const int*   __restrict__ labels,
                const bf16_t* __restrict__ meansbf,
                const float* __restrict__ msq,
                float* __restrict__ out_loss,
                float* __restrict__ out_preds,
                float* __restrict__ out_logits) {
  __shared__ bf16_t Asb[KSTEPS][BM * BSTRIDE]; // full 128x384 z block, bf16
  __shared__ bf16_t Bs[2][BN * BSTRIDE];       // double-buffered B tile
  __shared__ float  zsq[BM];
  __shared__ float  msqs[BN];
  __shared__ int    labs[BM];
  __shared__ float  loss_acc;

  const int tid  = threadIdx.x;
  const int row0 = blockIdx.x * BM;
  const int a    = blockIdx.y;

  if (tid < BM) {
    zsq[tid]  = 0.0f;
    labs[tid] = labels[(size_t)(row0 + tid) * NUM_ATTR + a];
  }
  if (tid == 0) loss_acc = 0.0f;
  __syncthreads();

  const int wave = tid >> 5;               // wave32
  const int lane = tid & 31;
  const int l16  = lane & 15;
  const int hi   = lane >> 4;
  const int wrow = wave * 16;

  // ---- Stage the entire A block once: z (f32, read from HBM exactly once)
  //      -> bf16 in LDS; accumulate |z|^2 per row in f32.
  {
    const float* zbase = z + (size_t)row0 * TOTAL_DIM + (size_t)a * KD;
    for (int ks = 0; ks < KSTEPS; ++ks) {
      #pragma unroll
      for (int it = 0; it < 4; ++it) {
        int idx = it * 256 + tid;          // 1024 float4 per chunk
        int r   = idx >> 3;                // row 0..127
        int k4  = (idx & 7) * 4;
        float4 v = *reinterpret_cast<const float4*>(
            zbase + (size_t)r * TOTAL_DIM + ks * TK + k4);
        cvt4_store(&Asb[ks][r * BSTRIDE + k4], v);
        atomicAdd(&zsq[r], v.x * v.x + v.y * v.y + v.z * v.z + v.w * v.w);
      }
    }
  }
  __syncthreads();                         // A block + zsq complete

  float runmax[8]; int runidx[8]; float selval[8];
  float zsq_r[8];  int lab_r[8];
  #pragma unroll
  for (int r = 0; r < 8; ++r) {
    runmax[r] = -3.4e38f; runidx[r] = 0; selval[r] = 0.0f;
    int rr = wrow + r + hi * 8;            // C-layout: VGPR r = rows r, r+8
    zsq_r[r] = zsq[rr];
    lab_r[r] = labs[rr];
  }

  const bf16_t* mbf_attr = meansbf + (size_t)a * NC * KD;

  for (int cb = 0; cb < NC / BN; ++cb) {
    __syncthreads();                       // prev epilogue done with msqs/Bs
    if (tid < BN) msqs[tid] = msq[(size_t)a * NC + cb * BN + tid];
    stage_B(mbf_attr, tid, Bs[0], cb, 0);

    v8f acc[8];
    #pragma unroll
    for (int t = 0; t < 8; ++t) acc[t] = (v8f){0, 0, 0, 0, 0, 0, 0, 0};

    int p = 0;
    for (int ks = 0; ks < KSTEPS; ++ks) {
      wait_async_staging();
      __syncthreads();                     // buf p ready; buf p^1 free
      if (ks + 1 < KSTEPS) stage_B(mbf_attr, tid, Bs[p ^ 1], cb, ks + 1);
      // A frag (ISA 7.12.2): lanes 0-15 -> K{0..7,16..23}, 16-31 -> K{8..15,24..31}
      v16bf af = frag_from_lds(&Asb[ks][(wrow + l16) * BSTRIDE + hi * 8]);
      // Batch all 8 B-fragment loads, then fire 8 back-to-back WMMAs.
      v16bf bf[8];
      #pragma unroll
      for (int t = 0; t < 8; ++t) {
        bf[t] = frag_from_lds(&Bs[p][(t * 16 + l16) * BSTRIDE + hi * 8]);
      }
      #pragma unroll
      for (int t = 0; t < 8; ++t) {
        acc[t] = __builtin_amdgcn_wmma_f32_16x16x32_bf16(
            false, af, false, bf[t], (short)0, acc[t], false, false);
      }
      p ^= 1;
    }

    // ---- fused epilogue for this col-block
    #pragma unroll
    for (int t = 0; t < 8; ++t) {
      int   col = cb * BN + t * 16 + l16;
      float mq  = msqs[t * 16 + l16];
      #pragma unroll
      for (int r = 0; r < 8; ++r) {
        float logit = 2.0f * acc[t][r] - zsq_r[r] - mq;
        int rr_g = row0 + wrow + r + hi * 8;
        __builtin_nontemporal_store(
            logit, &out_logits[(size_t)rr_g * (NUM_ATTR * NC) +
                               (size_t)a * NC + col]);
        if (logit > runmax[r]) { runmax[r] = logit; runidx[r] = col; }
        if (col == lab_r[r]) selval[r] += logit;   // exactly one hit per row
      }
    }
  }

  // ---- cross-lane argmax / label-gather within each 16-lane row group
  float wave_loss = 0.0f;
  #pragma unroll
  for (int r = 0; r < 8; ++r) {
    float m = runmax[r]; int ix = runidx[r]; float sv = selval[r];
    #pragma unroll
    for (int off = 8; off > 0; off >>= 1) {
      float om = __shfl_xor(m, off, 16);
      int   oi = __shfl_xor(ix, off, 16);
      float os = __shfl_xor(sv, off, 16);
      if (om > m || (om == m && oi < ix)) { m = om; ix = oi; }  // first index
      sv += os;
    }
    if (l16 == 0) {
      int rr_g = row0 + wrow + r + hi * 8;
      out_preds[(size_t)rr_g * NUM_ATTR + a] = (float)ix;
      wave_loss += -0.5f * sv;             // 0.5*|z-mu_lab|^2 = -0.5*logit_lab
    }
  }
  if (l16 == 0) atomicAdd(&loss_acc, wave_loss);

  if (a == 0 && tid < BM) {                // constant + sldj terms, once per row
    float extra = 0.5f * (float)TOTAL_DIM * LOG_2PI - sldj[row0 + tid];
    atomicAdd(&loss_acc, extra);
  }
  __syncthreads();
  if (tid == 0) atomicAdd(out_loss, loss_acc * (1.0f / (float)NS));
}

extern "C" void kernel_launch(void* const* d_in, const int* in_sizes, int n_in,
                              void* d_out, int out_size, void* d_ws, size_t ws_size,
                              hipStream_t stream) {
  (void)in_sizes; (void)n_in; (void)out_size; (void)ws_size;
  const float* z      = (const float*)d_in[0];
  const float* sldj   = (const float*)d_in[1];
  const int*   labels = (const int*)d_in[2];
  const float* means  = (const float*)d_in[3];

  float* out_loss   = (float*)d_out;
  float* out_preds  = out_loss + 1;
  float* out_logits = out_preds + (size_t)NS * NUM_ATTR;

  float*  msq     = (float*)d_ws;                               // 16 KB
  bf16_t* meansbf = (bf16_t*)((char*)d_ws + 4096 * sizeof(float)); // 3.1 MB

  gmm_init_kernel<<<1, 64, 0, stream>>>(out_loss);
  gmm_msq_kernel<<<(NUM_ATTR * NC) / 256, 256, 0, stream>>>(means, msq);
  gmm_cvt_means_kernel<<<(NUM_ATTR * NC * KD / 4) / 256, 256, 0, stream>>>(
      means, meansbf);

  dim3 grid(NS / BM, NUM_ATTR);            // (128, 8)
  gmm_main_kernel<<<grid, 256, 0, stream>>>(z, sldj, labels, meansbf, msq,
                                            out_loss, out_preds, out_logits);
}
